// ViewAggregator_89378269430315
// MI455X (gfx1250) — matrix operand order repaired and checked by reading, not compiled, not run on hardware
//
#include <hip/hip_runtime.h>
#include <hip/hip_bf16.h>
#include <math.h>

// ---------------------------------------------------------------------------
// ViewAggregator for MI455X (gfx1250): bf16 WMMA + async global->LDS staging
//   E=768, H=8, DH=96, B=8192, V=6
// ---------------------------------------------------------------------------

typedef __attribute__((ext_vector_type(16))) __bf16        v16bf;
typedef __attribute__((ext_vector_type(8)))  float         v8f;
typedef __attribute__((ext_vector_type(4)))  unsigned int  v4u;

union FragBf { v16bf bf; v4u u[2]; };

#define EDIM 768
#define HHEADS 8
#define DHEAD 96
#define BATCH 8192
#define NVIEW 6
#define NROWS (BATCH * NVIEW)   // 49152

// LDS-relative byte address of a pointer into a __shared__ object
#define LDS_OFF(p) ((unsigned)(uintptr_t)(__attribute__((address_space(3))) const void*)(p))

// async-counter waits (immediates)
#define WAIT_ASYNC_0() asm volatile("s_wait_asynccnt 0x0" ::: "memory")
#define WAIT_ASYNC_2() asm volatile("s_wait_asynccnt 0x2" ::: "memory")

__device__ __forceinline__ unsigned short f2bf(float f) {
    unsigned int u = __float_as_uint(f);
    u += 0x7FFFu + ((u >> 16) & 1u);       // round-to-nearest-even
    return (unsigned short)(u >> 16);
}

__device__ __forceinline__ v8f vzero8() {
    v8f z = {0.f, 0.f, 0.f, 0.f, 0.f, 0.f, 0.f, 0.f};
    return z;
}

// A-matrix fragment (16x32 bf16) from global, row-major rows of EDIM.
__device__ __forceinline__ v16bf ldA(const unsigned short* A, long rowbase, int k0, int lane) {
    const unsigned short* p = A + (rowbase + (lane & 15)) * (long)EDIM + k0 + ((lane >> 4) << 3);
    FragBf f;
    f.u[0] = *(const v4u*)(p);
    f.u[1] = *(const v4u*)(p + 16);
    return f.bf;
}

// B-matrix fragment (32x16 bf16) from an LDS panel laid out [col][32k].
__device__ __forceinline__ v16bf ldB_lds(const unsigned short* pan, int n0, int lane) {
    const unsigned short* p = pan + (n0 + (lane & 15)) * 32 + ((lane >> 4) << 3);
    FragBf f;
    f.u[0] = *(const v4u*)(p);
    f.u[1] = *(const v4u*)(p + 16);
    return f.bf;
}

__device__ __forceinline__ void async_chunk(unsigned short* lds_dst,
                                            const unsigned short* gbase,
                                            int col0, int k0, int c) {
    int col = c >> 2;                      // panel column
    int ch  = c & 3;                       // 16B chunk within the 64B row
    unsigned lds_off = LDS_OFF(lds_dst + col * 32 + ch * 8);
    unsigned goff    = (unsigned)((((long)(col0 + col)) * EDIM + k0) * 2 + ch * 16);
    asm volatile("global_load_async_to_lds_b128 %0, %1, %2"
                 :: "v"(lds_off), "v"(goff), "s"(gbase)
                 : "memory");
}

// Stage a weight panel [ncols x 32k] into LDS. REQUIRES nthreads == ncols*2:
// each thread issues exactly 2 async ops (straight-line, no divergence), so
// every wave's ASYNCcnt rises by exactly 2 per stage.
__device__ __forceinline__ void stage_panel(unsigned short* lds_dst,
                                            const unsigned short* gbase,
                                            int col0, int k0,
                                            int tid, int nthreads) {
    async_chunk(lds_dst, gbase, col0, k0, tid);
    async_chunk(lds_dst, gbase, col0, k0, tid + nthreads);
}

// ---------------------------------------------------------------------------
// Kernel 1: fp32 -> bf16 weight staging (wk;wv contiguous, and out_w)
// ---------------------------------------------------------------------------
__global__ void prep_weights(const float* __restrict__ in_proj_w,
                             const float* __restrict__ out_w,
                             unsigned short* __restrict__ wkv,
                             unsigned short* __restrict__ wout) {
    long i = (long)blockIdx.x * blockDim.x + threadIdx.x;
    long stride = (long)gridDim.x * blockDim.x;
    const long NKV = 2L * EDIM * EDIM;   // wk (768x768) then wv (768x768)
    const long NO  = (long)EDIM * EDIM;
    for (long t = i; t < NKV; t += stride)
        wkv[t] = f2bf(in_proj_w[(long)EDIM * EDIM + t]);
    for (long t = i; t < NO; t += stride)
        wout[t] = f2bf(out_w[t]);
}

// ---------------------------------------------------------------------------
// Kernel 2: q = (query @ wq^T + bq) * (1/sqrt(96)), fp32, 768 values
// ---------------------------------------------------------------------------
__global__ void make_q(const float* __restrict__ query,
                       const float* __restrict__ in_proj_w,
                       const float* __restrict__ in_proj_b,
                       float* __restrict__ qv) {
    int o = blockIdx.x * blockDim.x + threadIdx.x;
    if (o >= EDIM) return;
    float s = in_proj_b[o];
    const float* wr = in_proj_w + (long)o * EDIM;
    for (int k = 0; k < EDIM; ++k) s += query[k] * wr[k];
    qv[o] = s * 0.10206207261596575f;    // 1/sqrt(96)
}

// ---------------------------------------------------------------------------
// Kernel 3: pos-MLP + kv = view_features + pos, store bf16 (49152 x 768)
// ---------------------------------------------------------------------------
__global__ __launch_bounds__(256) void build_kv(const float* __restrict__ feat,
                                                const float* __restrict__ ang,
                                                const float* __restrict__ w1,
                                                const float* __restrict__ b1,
                                                const float* __restrict__ w2,
                                                const float* __restrict__ b2,
                                                unsigned short* __restrict__ kvA) {
    __shared__ float hid[128];
    long row = blockIdx.x;
    float a0 = ang[row * 2 + 0];
    float a1 = ang[row * 2 + 1];
    int t = threadIdx.x;
    if (t < 128) {
        float hv = a0 * w1[t * 2] + a1 * w1[t * 2 + 1] + b1[t];
        hid[t] = hv > 0.f ? hv : 0.f;
    }
    __syncthreads();
    for (int o = t; o < EDIM; o += blockDim.x) {
        float s = b2[o];
        const float* wr = w2 + (long)o * 128;
        #pragma unroll 4
        for (int j = 0; j < 128; ++j) s += hid[j] * wr[j];
        kvA[row * (long)EDIM + o] = f2bf(feat[row * (long)EDIM + o] + s);
    }
}

// ---------------------------------------------------------------------------
// Kernel 4: fused K/V projection (bf16 WMMA) + per-head softmax + ctx.
// Block = 16 batch elems = 96 kv rows; 6 waves, one 16-row strip per wave.
// Weight panels (96 cols x 32 k) double-buffered into LDS via async DMA.
// ---------------------------------------------------------------------------
__global__ __launch_bounds__(192) void attn_fused(const unsigned short* __restrict__ kvA,
                                                  const unsigned short* __restrict__ wkv,
                                                  const float* __restrict__ qv,
                                                  const float* __restrict__ in_proj_b,
                                                  unsigned short* __restrict__ ctx) {
    __shared__ float sc_s[96];
    __shared__ float at_s[96];
    __shared__ float cx_s[16 * 96];
    __shared__ unsigned short bpan[2][96 * 32];   // double-buffered B panel
    int tid  = threadIdx.x;
    int w    = tid >> 5;
    int lane = tid & 31;
    long rowbase = (long)blockIdx.x * 96 + (long)w * 16;
    long bbase   = (long)blockIdx.x * 16;
    int half = (lane >> 4) * 8;

    for (int h = 0; h < HHEADS; ++h) {
        for (int proj = 0; proj < 2; ++proj) {    // 0 = K, 1 = V
            int wcol0 = proj * EDIM + h * DHEAD;  // row offset into wkv
            v8f acc[6];
            #pragma unroll
            for (int n = 0; n < 6; ++n) acc[n] = vzero8();

            stage_panel(bpan[0], wkv, wcol0, 0, tid, 192);
            for (int kt = 0; kt < 24; ++kt) {
                int cur = kt & 1;
                if (kt < 23) {
                    stage_panel(bpan[cur ^ 1], wkv, wcol0, (kt + 1) * 32, tid, 192);
                    WAIT_ASYNC_2();               // retire current panel, keep next in flight
                } else {
                    WAIT_ASYNC_0();
                }
                __syncthreads();                  // panel visible to all waves
                int k0 = kt * 32;
                v16bf a = ldA(kvA, rowbase, k0, lane);
                #pragma unroll
                for (int n = 0; n < 6; ++n) {
                    v16bf b = ldB_lds(bpan[cur], n * 16, lane);
                    acc[n] = __builtin_amdgcn_wmma_f32_16x16x32_bf16(
                        false, a, false, b, (short)0, acc[n], false, false);
                }
                __syncthreads();                  // all reads done before buffer reuse
            }

            if (proj == 0) {
                // scores: s[row] = sum_col (k[row,col]+bk[col]) * qscaled[col]
                float sc[8];
                #pragma unroll
                for (int r = 0; r < 8; ++r) sc[r] = 0.f;
                #pragma unroll
                for (int n = 0; n < 6; ++n) {
                    int col = h * DHEAD + n * 16 + (lane & 15);
                    float q  = qv[col];
                    float bk = in_proj_b[EDIM + col];
                    #pragma unroll
                    for (int r = 0; r < 8; ++r) {
                        float vsum = (acc[n][r] + bk) * q;
                        vsum += __shfl_xor(vsum, 1, 32);
                        vsum += __shfl_xor(vsum, 2, 32);
                        vsum += __shfl_xor(vsum, 4, 32);
                        vsum += __shfl_xor(vsum, 8, 32);   // within 16-lane half
                        sc[r] += vsum;
                    }
                }
                if ((lane & 15) == 0) {
                    #pragma unroll
                    for (int r = 0; r < 8; ++r) sc_s[w * 16 + half + r] = sc[r];
                }
                __syncthreads();
                // softmax over 6 views per batch elem; zero ctx accumulator
                if (tid < 16) {
                    float m = -1e30f;
                    for (int v = 0; v < 6; ++v) { float s = sc_s[tid * 6 + v]; m = s > m ? s : m; }
                    float e[6]; float sum = 0.f;
                    for (int v = 0; v < 6; ++v) { float ee = __expf(sc_s[tid * 6 + v] - m); e[v] = ee; sum += ee; }
                    float inv = 1.f / sum;
                    for (int v = 0; v < 6; ++v) at_s[tid * 6 + v] = e[v] * inv;
                }
                for (int i = tid; i < 16 * 96; i += 192) cx_s[i] = 0.f;
                __syncthreads();
            } else {
                // ctx[b, col] += attn[row] * (v[row,col] + bv[col])
                #pragma unroll
                for (int n = 0; n < 6; ++n) {
                    int col = n * 16 + (lane & 15);
                    float bv = in_proj_b[2 * EDIM + h * DHEAD + col];
                    #pragma unroll
                    for (int r = 0; r < 8; ++r) {
                        int mloc = w * 16 + half + r;
                        float val = (acc[n][r] + bv) * at_s[mloc];
                        atomicAdd(&cx_s[(mloc / 6) * 96 + col], val);
                    }
                }
                __syncthreads();
                for (int i = tid; i < 16 * 96; i += 192) {
                    int bl = i / 96, c = i % 96;
                    ctx[(bbase + bl) * (long)EDIM + h * DHEAD + c] = f2bf(cx_s[i]);
                }
                __syncthreads();
            }
        }
    }
}

// ---------------------------------------------------------------------------
// Kernel 5: out = ctx(bf16) @ out_w^T(bf16) + out_b, fp32 result.
// Block = 128 rows (8 waves); weight panels (128 x 32) async-staged in LDS.
// ---------------------------------------------------------------------------
__global__ __launch_bounds__(256) void out_gemm(const unsigned short* __restrict__ ctx,
                                                const unsigned short* __restrict__ wout,
                                                const float* __restrict__ out_b,
                                                float* __restrict__ out) {
    __shared__ unsigned short bpan[2][128 * 32];
    int tid  = threadIdx.x;
    int w    = tid >> 5;
    int lane = tid & 31;
    long rowbase = (long)blockIdx.x * 128 + (long)w * 16;
    int half = (lane >> 4) * 8;

    for (int nc = 0; nc < 6; ++nc) {
        __builtin_prefetch((const void*)(ctx + rowbase * EDIM), 0, 3);
        v8f acc[8];
        #pragma unroll
        for (int n = 0; n < 8; ++n) acc[n] = vzero8();

        stage_panel(bpan[0], wout, nc * 128, 0, tid, 256);
        for (int kt = 0; kt < 24; ++kt) {
            int cur = kt & 1;
            if (kt < 23) {
                stage_panel(bpan[cur ^ 1], wout, nc * 128, (kt + 1) * 32, tid, 256);
                WAIT_ASYNC_2();
            } else {
                WAIT_ASYNC_0();
            }
            __syncthreads();
            int k0 = kt * 32;
            v16bf a = ldA(ctx, rowbase, k0, lane);
            #pragma unroll
            for (int n = 0; n < 8; ++n) {
                v16bf b = ldB_lds(bpan[cur], n * 16, lane);
                acc[n] = __builtin_amdgcn_wmma_f32_16x16x32_bf16(
                    false, a, false, b, (short)0, acc[n], false, false);
            }
            __syncthreads();
        }
        #pragma unroll
        for (int n = 0; n < 8; ++n) {
            int col = nc * 128 + n * 16 + (lane & 15);
            float bias = out_b[col];
            #pragma unroll
            for (int r = 0; r < 8; ++r) {
                out[(rowbase + half + r) * (long)EDIM + col] = acc[n][r] + bias;
            }
        }
    }
}

// ---------------------------------------------------------------------------
// Host-side launcher
// ---------------------------------------------------------------------------
extern "C" void kernel_launch(void* const* d_in, const int* in_sizes, int n_in,
                              void* d_out, int out_size, void* d_ws, size_t ws_size,
                              hipStream_t stream) {
    const float* feat  = (const float*)d_in[0];   // (8192, 6, 768)
    const float* ang   = (const float*)d_in[1];   // (8192, 6, 2)
    const float* w1    = (const float*)d_in[2];   // (128, 2)
    const float* b1    = (const float*)d_in[3];   // (128,)
    const float* w2    = (const float*)d_in[4];   // (768, 128)
    const float* b2    = (const float*)d_in[5];   // (768,)
    const float* query = (const float*)d_in[6];   // (1, 1, 768)
    const float* ipw   = (const float*)d_in[7];   // (2304, 768)
    const float* ipb   = (const float*)d_in[8];   // (2304,)
    const float* outw  = (const float*)d_in[9];   // (768, 768)
    const float* outb  = (const float*)d_in[10];  // (768,)
    float* out = (float*)d_out;                   // (8192, 768) f32

    char* ws = (char*)d_ws;
    float*          qv   = (float*)(ws);
    unsigned short* wkv  = (unsigned short*)(ws + 3072);
    unsigned short* wout = (unsigned short*)(ws + 3072 + 2359296);
    unsigned short* kvA  = (unsigned short*)(ws + 3072 + 2359296 + 1179648);
    unsigned short* ctx  = (unsigned short*)(ws + 3072 + 2359296 + 1179648 + 75497472);

    prep_weights<<<2048, 256, 0, stream>>>(ipw, outw, wkv, wout);
    make_q<<<3, 256, 0, stream>>>(query, ipw, ipb, qv);
    build_kv<<<NROWS, 256, 0, stream>>>(feat, ang, w1, b1, w2, b2, kvA);
    attn_fused<<<BATCH / 16, 192, 0, stream>>>(kvA, wkv, qv, ipb, ctx);
    out_gemm<<<BATCH / 128, 256, 0, stream>>>(ctx, wout, outb, out);
}
